// VQCModel_4767413699141
// MI455X (gfx1250) — compile-verified, matrix operand order checked by reading, not run on hardware
//
#include <hip/hip_runtime.h>
#include <hip/hip_bf16.h>
#include <math.h>

// ---------------------------------------------------------------------------
// VQC: 8 qubits, 2 layers, batch 4096.
// Kernel 1: one wave32 per batch element; 256-amp complex state held entirely
//           in registers (8 complex amps per lane). Cross-lane qubits via
//           __shfl_xor, local qubits via register pairs. Fully unrolled.
// Kernel 2: linear head ev(4096x8) @ W^T(8x10) + b via V_WMMA_F32_16X16X4_F32
//           (two K=4 steps accumulate K=8), one wave per 16-row tile.
// ---------------------------------------------------------------------------

typedef __attribute__((ext_vector_type(2)))  float    v2f;
typedef __attribute__((ext_vector_type(8)))  float    v8f;
typedef __attribute__((ext_vector_type(16))) _Float16 v16h;

struct cplx { float re, im; };

__device__ __forceinline__ float shx(float v, int mask) {
  return __shfl_xor(v, mask, 32);
}

// r = u*a + v*b (complex)
__device__ __forceinline__ cplx cmul2(cplx u, cplx a, cplx v, cplx b) {
  cplx r;
  r.re = u.re*a.re - u.im*a.im + v.re*b.re - v.im*b.im;
  r.im = u.re*a.im + u.im*a.re + v.re*b.im + v.im*b.re;
  return r;
}

// Generic 1q gate on flat-index bit position p (0..7).  s = lane | (k<<5).
__device__ __forceinline__ void apply_gate(cplx amp[8], int lane, int p,
                                           cplx u00, cplx u01, cplx u10, cplx u11) {
  if (p >= 5) {                 // register-local bit
    const int m = 1 << (p - 5);
#pragma unroll
    for (int k = 0; k < 8; ++k) {
      if ((k & m) == 0) {
        cplx a0 = amp[k], a1 = amp[k | m];
        amp[k]     = cmul2(u00, a0, u01, a1);
        amp[k | m] = cmul2(u10, a0, u11, a1);
      }
    }
  } else {                      // lane bit: pair via shuffle, branch-free select
    const int lm = 1 << p;
    const bool hi = (lane & lm) != 0;
#pragma unroll
    for (int k = 0; k < 8; ++k) {
      cplx part; part.re = shx(amp[k].re, lm); part.im = shx(amp[k].im, lm);
      cplx mine = amp[k];
      cplx rlo = cmul2(u00, mine, u01, part);   // this lane holds a0
      cplx rhi = cmul2(u10, part, u11, mine);   // this lane holds a1
      amp[k].re = hi ? rhi.re : rlo.re;
      amp[k].im = hi ? rhi.im : rlo.im;
    }
  }
}

// RZ(theta): diag(e^{-i t}, e^{+i t}) with t = theta/2 — pure phase, no pairing.
__device__ __forceinline__ void apply_rz(cplx amp[8], int lane, int p, float half) {
  float c = cosf(half), s = sinf(half);
#pragma unroll
  for (int k = 0; k < 8; ++k) {
    int bit = (p >= 5) ? ((k >> (p - 5)) & 1) : ((lane >> p) & 1);
    float si = bit ? s : -s;                 // e^{+i t} for |1>, e^{-i t} for |0>
    cplx a = amp[k];
    amp[k].re = a.re * c - a.im * si;
    amp[k].im = a.im * c + a.re * si;
  }
}

// CNOT: for indices with control bit pc==1, flip target bit pt.
__device__ __forceinline__ void apply_cnot(cplx amp[8], int lane, int pc, int pt) {
  if (pc >= 5 && pt >= 5) {                       // both register-local
    const int cm = 1 << (pc - 5), tm = 1 << (pt - 5);
#pragma unroll
    for (int k = 0; k < 8; ++k)
      if ((k & cm) && !(k & tm)) {
        cplx t = amp[k]; amp[k] = amp[k | tm]; amp[k | tm] = t;
      }
  } else if (pc >= 5) {                           // control local, target on lanes
    const int cm = 1 << (pc - 5), lm = 1 << pt;
#pragma unroll
    for (int k = 0; k < 8; ++k)
      if (k & cm) {                               // wave-uniform branch (depends on k)
        amp[k].re = shx(amp[k].re, lm);
        amp[k].im = shx(amp[k].im, lm);
      }
  } else if (pt >= 5) {                           // control on lanes, target local
    const int tm = 1 << (pt - 5);
    const bool ctrl = ((lane >> pc) & 1) != 0;
#pragma unroll
    for (int k = 0; k < 8; ++k)
      if (!(k & tm)) {
        cplx a = amp[k], b = amp[k | tm];
        amp[k].re      = ctrl ? b.re : a.re;  amp[k].im      = ctrl ? b.im : a.im;
        amp[k | tm].re = ctrl ? a.re : b.re;  amp[k | tm].im = ctrl ? a.im : b.im;
      }
  } else {                                        // both on lanes
    const int lm = 1 << pt;
    const bool ctrl = ((lane >> pc) & 1) != 0;
#pragma unroll
    for (int k = 0; k < 8; ++k) {
      float tr = shx(amp[k].re, lm), ti = shx(amp[k].im, lm);
      amp[k].re = ctrl ? tr : amp[k].re;
      amp[k].im = ctrl ? ti : amp[k].im;
    }
  }
}

__global__ __launch_bounds__(256)
void vqc_state_kernel(const float* __restrict__ x, const float* __restrict__ params,
                      float* __restrict__ ev_out, int B) {
  const int lane = threadIdx.x & 31;
  const int wave = (int)((blockIdx.x * blockDim.x + threadIdx.x) >> 5);
  if (wave >= B) return;

  cplx amp[8];
#pragma unroll
  for (int k = 0; k < 8; ++k) { amp[k].re = 0.0f; amp[k].im = 0.0f; }
  amp[0].re = (lane == 0) ? 1.0f : 0.0f;          // |0...0>

  // --- data-encoding RY(x_i * pi) on qubit i  (qubit w lives at bit 7-w) ---
#pragma unroll
  for (int w = 0; w < 8; ++w) {
    float th = x[wave * 8 + w] * (3.14159265358979323846f * 0.5f);  // half angle
    float c = cosf(th), s = sinf(th);
    apply_gate(amp, lane, 7 - w,
               cplx{c, 0.0f}, cplx{-s, 0.0f}, cplx{s, 0.0f}, cplx{c, 0.0f});
  }

  // --- layers: RX,RY,RZ per qubit then CNOT+RZ entanglers ---
#pragma unroll
  for (int l = 0; l < 2; ++l) {
    const int base = l * 52;                      // 24 rot + 28 ent per layer
#pragma unroll
    for (int i = 0; i < 8; ++i) {
      const int p = 7 - i;
      float tx = params[base + 3 * i + 0] * 0.5f;
      float cx = cosf(tx), sx = sinf(tx);
      apply_gate(amp, lane, p, cplx{cx, 0.0f}, cplx{0.0f, -sx},
                               cplx{0.0f, -sx}, cplx{cx, 0.0f});    // RX
      float ty = params[base + 3 * i + 1] * 0.5f;
      float cy = cosf(ty), sy = sinf(ty);
      apply_gate(amp, lane, p, cplx{cy, 0.0f}, cplx{-sy, 0.0f},
                               cplx{sy, 0.0f}, cplx{cy, 0.0f});     // RY
      apply_rz(amp, lane, p, params[base + 3 * i + 2] * 0.5f);      // RZ
    }
    int kk = base + 24;
#pragma unroll
    for (int i = 0; i < 8; ++i) {
#pragma unroll
      for (int j = i + 1; j < 8; ++j) {
        apply_cnot(amp, lane, 7 - i, 7 - j);
        apply_rz(amp, lane, 7 - j, params[kk] * 0.5f);
        ++kk;
      }
    }
  }

  // --- <Z_w> = sum |amp|^2 * (1 - 2*bit_{7-w}(s)) ---
  float ptot = 0.0f, d0 = 0.0f, d1 = 0.0f, d2 = 0.0f;
#pragma unroll
  for (int k = 0; k < 8; ++k) {
    float pr = amp[k].re * amp[k].re + amp[k].im * amp[k].im;
    ptot += pr;
    d0 += (k & 1) ? -pr : pr;
    d1 += (k & 2) ? -pr : pr;
    d2 += (k & 4) ? -pr : pr;
  }
  float evl[8];
  evl[0] = d2;  evl[1] = d1;  evl[2] = d0;        // qubits 0..2 -> k bits 2..0
#pragma unroll
  for (int w = 3; w < 8; ++w) {
    int p = 7 - w;                                // lane bit
    evl[w] = ((lane >> p) & 1) ? -ptot : ptot;
  }
#pragma unroll
  for (int w = 0; w < 8; ++w) {
#pragma unroll
    for (int m = 1; m < 32; m <<= 1) evl[w] += shx(evl[w], m);
  }
  if (lane == 0) {
#pragma unroll
    for (int w = 0; w < 8; ++w) ev_out[wave * 8 + w] = evl[w];
  }
}

// ---------------------------------------------------------------------------
// Linear head: out(4096x10) = ev(4096x8) @ W^T + b, 16 rows per wave via WMMA.
// ---------------------------------------------------------------------------
#if __has_builtin(__builtin_amdgcn_wmma_f32_16x16x4_f32)
#define HAVE_WMMA_F32X4 1
#else
#define HAVE_WMMA_F32X4 0
#endif

__global__ __launch_bounds__(256)
void vqc_head_kernel(const float* __restrict__ ev, const float* __restrict__ W,
                     const float* __restrict__ bias, float* __restrict__ out, int B) {
  const int lane = threadIdx.x & 31;
  const int tile = (int)((blockIdx.x * blockDim.x + threadIdx.x) >> 5);
  const int row0 = tile * 16;
  if (row0 >= B) return;                           // wave-uniform

  const int m  = lane & 15;                        // M (A rows) / N (B cols)
  const int kh = lane >> 4;                        // K-pair selector
  const float  wsel = (m < 10) ? 1.0f : 0.0f;      // zero-pad N to 16
  const int    n    = (m < 10) ? m : 0;
  const float* arow = ev + (size_t)(row0 + m) * 8;
  const float* wrow = W + (size_t)n * 8;           // B[k][n] = W[n][k]

  v8f c = {0.f, 0.f, 0.f, 0.f, 0.f, 0.f, 0.f, 0.f};

#if HAVE_WMMA_F32X4
  // ISA 32-bit 16x4 layout: lanes 0-15 hold K={0,1}, lanes 16-31 K={2,3}.
  v2f a0, a1, b0, b1;
  a0.x = arow[2 * kh];     a0.y = arow[2 * kh + 1];
  a1.x = arow[4 + 2 * kh]; a1.y = arow[5 + 2 * kh];
  b0.x = wrow[2 * kh] * wsel;     b0.y = wrow[2 * kh + 1] * wsel;
  b1.x = wrow[4 + 2 * kh] * wsel; b1.y = wrow[5 + 2 * kh] * wsel;
  c = __builtin_amdgcn_wmma_f32_16x16x4_f32(false, a0, false, b0,
                                            (short)0, c, false, false);
  c = __builtin_amdgcn_wmma_f32_16x16x4_f32(false, a1, false, b1,
                                            (short)0, c, false, false);
#else
  // Fallback: probe-confirmed f16 WMMA, K padded 8 -> 32 with zeros.
  // Only lanes 0-15, elements 0..7 carry K=0..7 for both A and B; all K>=8
  // slots are zero so either candidate B layout yields the same product.
  v16h a, b;
#pragma unroll
  for (int e = 0; e < 16; ++e) { a[e] = (_Float16)0.0f; b[e] = (_Float16)0.0f; }
  if (kh == 0) {
#pragma unroll
    for (int e = 0; e < 8; ++e) {
      a[e] = (_Float16)arow[e];
      b[e] = (_Float16)(wrow[e] * wsel);
    }
  }
  c = __builtin_amdgcn_wmma_f32_16x16x32_f16(false, a, false, b,
                                             (short)0, c, false, false);
#endif

  // C/D layout: VGPR r -> row r + 8*kh, col = lane&15.
  float bv = (m < 10) ? bias[m] : 0.0f;
#pragma unroll
  for (int r = 0; r < 8; ++r) {
    int row = row0 + r + 8 * kh;
    if (m < 10 && row < B) out[(size_t)row * 10 + m] = c[r] + bv;
  }
}

extern "C" void kernel_launch(void* const* d_in, const int* in_sizes, int n_in,
                              void* d_out, int out_size, void* d_ws, size_t ws_size,
                              hipStream_t stream) {
  const float* x      = (const float*)d_in[0];   // (B, 8)
  const float* params = (const float*)d_in[1];   // (104,)
  const float* W      = (const float*)d_in[2];   // (10, 8)
  const float* bias   = (const float*)d_in[3];   // (10,)
  float* out = (float*)d_out;                    // (B, 10)
  float* ev  = (float*)d_ws;                     // (B, 8) scratch, 128 KB

  const int B = in_sizes[0] / 8;                 // 4096

  // Kernel 1: one wave32 per batch element (8 waves / 256-thread block).
  {
    const int threads = 256;
    const int blocks  = (B * 32 + threads - 1) / threads;
    vqc_state_kernel<<<blocks, threads, 0, stream>>>(x, params, ev, B);
  }
  // Kernel 2: one wave32 per 16-row output tile.
  {
    const int tiles   = (B + 15) / 16;
    const int threads = 256;
    const int blocks  = (tiles * 32 + threads - 1) / threads;
    vqc_head_kernel<<<blocks, threads, 0, stream>>>(ev, W, bias, out, B);
  }
}
